// FlashMTPAttention_58145267253995
// MI455X (gfx1250) — matrix operand order repaired and checked by reading, not compiled, not run on hardware
//
#include <hip/hip_runtime.h>

// Problem constants (match reference)
#define Bb    8
#define QQ    128
#define CTXN  4096
#define HH    2048
#define NHH   16
#define NKVH  8
#define HDD   128
#define SS    (CTXN + QQ)     // 4224
#define EPSF  1e-6f

typedef __attribute__((ext_vector_type(16))) __bf16 v16bf;
typedef __attribute__((ext_vector_type(8)))  float  v8f;
typedef __attribute__((ext_vector_type(2)))  __bf16 bf16x2;
typedef __attribute__((ext_vector_type(8)))  __bf16 bf16x8;
typedef __attribute__((ext_vector_type(4)))  unsigned tdm_u4;
typedef __attribute__((ext_vector_type(8)))  int      tdm_i8;
typedef __attribute__((ext_vector_type(4)))  int      tdm_i4;

#define LDA 72    // GEMM LDS tile ld: 144B rows -> 16B aligned, 36-bank shift
#define KTT 64    // attention key tile
#define KLD 136   // K tile ld: 272B rows -> 16B aligned, 4-bank row shift
#define VLD 128   // V tile ld (TDM writes unpadded [key][hd])
#define PLD 68    // P tile ld (overlaid on K buffer: 128*68 == 64*136)

// ---------------------------------------------------------------------------
// WMMA wrapper: D = A(16x32 bf16) * B(32x16 bf16) + C(16x16 f32)
// ---------------------------------------------------------------------------
__device__ __forceinline__ v8f wmma_bf16(v16bf a, v16bf b, v8f c) {
  return __builtin_amdgcn_wmma_f32_16x16x32_bf16(false, a, false, b, (short)0, c,
                                                 false, false);
}

__device__ __forceinline__ v8f zero8() {
  v8f z;
#pragma unroll
  for (int i = 0; i < 8; ++i) z[i] = 0.0f;
  return z;
}

// ---------------------------------------------------------------------------
// gfx1250 async global->LDS copy (ASYNCcnt path, ISA 15.18.3 op 98).
// ---------------------------------------------------------------------------
__device__ __forceinline__ void async_g2l_b128(const __bf16* g, __bf16* l) {
  unsigned laddr = (unsigned)(unsigned long long)(l);
  asm volatile("global_load_async_to_lds_b128 %0, %1, off"
               :: "v"(laddr), "v"(g)
               : "memory");
}

// Copy one KTT x 128 bf16 tile (16 KB) with 4 async b128 chunks per thread.
__device__ __forceinline__ void async_tile(const __bf16* __restrict__ gbase,
                                           __bf16* lbase, int ldl) {
  const int tid = threadIdx.x;
#pragma unroll
  for (int i = 0; i < 4; ++i) {
    const int idx = tid + 256 * i;   // 0..1023 chunks of 8 bf16 (16B)
    const int row = idx >> 4;
    const int c8 = (idx & 15) * 8;
    async_g2l_b128(gbase + (size_t)row * HDD + c8, lbase + row * ldl + c8);
  }
}

// ---------------------------------------------------------------------------
// Tensor Data Mover: DMA a KTT x 128 bf16 2D tile global->LDS (TENSORcnt).
// D# packing per CDNA5 ISA:
//   group0: [0] count=1 | [1] lds_addr | [2] gaddr[31:0] | [3] gaddr[56:32]|type2
//   group1: [0] data_size=2B | [1] tensor_dim0=128<<16 | [2] tensor_dim1 lo<<16
//           [3] tile_dim0=128<<16 | [4] tile_dim1=64 | [5] dim0_stride=128
//   groups 2/3 zero (2D tile).
// ---------------------------------------------------------------------------
__device__ __forceinline__ void tdm_load_tile(const __bf16* g, __bf16* l) {
  const unsigned long long ga = (unsigned long long)g;
  tdm_u4 g0;
  g0[0] = 1u;                                  // count=1, user descriptor
  g0[1] = (unsigned)(unsigned long long)l;     // lds byte address
  g0[2] = (unsigned)ga;                        // global_addr[31:0]
  g0[3] = (unsigned)(ga >> 32) | (2u << 30);   // global_addr[56:32] | type=2
  tdm_i8 g1;
  g1[0] = (int)(1u << 16);                     // data_size=2B; no pad/multicast
  g1[1] = (int)(128u << 16);                   // tensor_dim0 = 128
  g1[2] = (int)(0xFFFFu << 16);                // tensor_dim1 = 65535 (no OOB)
  g1[3] = (int)(128u << 16);                   // tile_dim0 = 128
  g1[4] = 64;                                  // tile_dim1 = 64, tile_dim2 = 0
  g1[5] = 128;                                 // tensor_dim0_stride = 128
  g1[6] = 0;
  g1[7] = 0;
  tdm_i4 gz;
  gz[0] = 0; gz[1] = 0; gz[2] = 0; gz[3] = 0;
#if __clang_major__ >= 23
  tdm_i8 gz8;
#pragma unroll
  for (int i = 0; i < 8; ++i) gz8[i] = 0;
  __builtin_amdgcn_tensor_load_to_lds(g0, g1, gz, gz, gz8, 0);
#else
  __builtin_amdgcn_tensor_load_to_lds(g0, g1, gz, gz, 0);
#endif
}

// ---------------------------------------------------------------------------
// Fragment loaders (layouts per CDNA5 ISA §7.12.2, wave32)
// ---------------------------------------------------------------------------
__device__ __forceinline__ v16bf load_frag_a(const __bf16* lds, int ld) {
  const int lane = threadIdx.x & 31;
  const int m = lane & 15, half = lane >> 4;
  const __bf16* row = lds + m * ld + 8 * half;
  v16bf a;
#pragma unroll
  for (int j = 0; j < 8; ++j) {
    const int kb = (j < 4) ? (2 * j) : (2 * j + 8);
    bf16x2 p = *(const bf16x2*)(row + kb);
    a[2 * j] = p.x;
    a[2 * j + 1] = p.y;
  }
  return a;
}

// B 32x16 (KxN), data stored [N][K]: B[k][n] = lds[n*ld + k]
__device__ __forceinline__ v16bf load_frag_b_nk(const __bf16* lds, int ld) {
  const int lane = threadIdx.x & 31;
  const int n = lane & 15, half = lane >> 4;
  const __bf16* row = lds + n * ld + 8 * half;
  v16bf b;
#pragma unroll
  for (int j = 0; j < 8; ++j) {
    const int kb = (j < 4) ? (2 * j) : (2 * j + 8);
    bf16x2 p = *(const bf16x2*)(row + kb);
    b[2 * j] = p.x;
    b[2 * j + 1] = p.y;
  }
  return b;
}

// B 32x16 (KxN), data stored [K][N]: B[k][n] = lds[k*ld + n]
__device__ __forceinline__ v16bf load_frag_b_kn(const __bf16* lds, int ld) {
  const int lane = threadIdx.x & 31;
  const int n = lane & 15, half = lane >> 4;
  const __bf16* col = lds + n + (8 * half) * ld;
  v16bf b;
#pragma unroll
  for (int j = 0; j < 8; ++j) {
    const int kb = (j < 4) ? (2 * j) : (2 * j + 8);
    b[2 * j] = col[kb * ld];
    b[2 * j + 1] = col[(kb + 1) * ld];
  }
  return b;
}

// ---------------------------------------------------------------------------
// GEMM staging, register-pipelined.
// ---------------------------------------------------------------------------
__device__ __forceinline__ void stage_load(const float* __restrict__ g, int ldg,
                                           float4 t[8]) {
  const int tid = threadIdx.x;
#pragma unroll
  for (int i = 0; i < 4; ++i) {
    const int idx = tid + 256 * i;   // 0..1023 chunks of 8 floats
    const int row = idx >> 3;
    const int c8 = (idx & 7) * 8;
    const float4* p = (const float4*)(g + (size_t)row * ldg + c8);
    t[2 * i] = p[0];
    t[2 * i + 1] = p[1];
  }
}

__device__ __forceinline__ void stage_store(const float4 t[8], __bf16* lds) {
  const int tid = threadIdx.x;
#pragma unroll
  for (int i = 0; i < 4; ++i) {
    const int idx = tid + 256 * i;
    const int row = idx >> 3;
    const int c8 = (idx & 7) * 8;
    bf16x8 o;
    o[0] = (__bf16)t[2 * i].x;     o[1] = (__bf16)t[2 * i].y;
    o[2] = (__bf16)t[2 * i].z;     o[3] = (__bf16)t[2 * i].w;
    o[4] = (__bf16)t[2 * i + 1].x; o[5] = (__bf16)t[2 * i + 1].y;
    o[6] = (__bf16)t[2 * i + 1].z; o[7] = (__bf16)t[2 * i + 1].w;
    *(bf16x8*)(lds + row * LDA + c8) = o;   // ds_store_b128
  }
}

// ---------------------------------------------------------------------------
// GEMM main loop with global-load pipelining and B-fragment prefetch.
// ---------------------------------------------------------------------------
__device__ __forceinline__ void gemm_mainloop(const float* __restrict__ X, int ldx,
                                              const float* __restrict__ W, int ldw,
                                              int ktot, __bf16* At, __bf16* Wt,
                                              v8f acc[8]) {
  const int wave = threadIdx.x >> 5;
  float4 tx[8], tw[8];
  stage_load(X, ldx, tx);
  stage_load(W, ldw, tw);
  for (int k0 = 0; k0 < ktot; k0 += 64) {
    stage_store(tx, At);
    stage_store(tw, Wt);
    __syncthreads();
    if (k0 + 64 < ktot) {
      __builtin_prefetch(X + k0 + 128, 0, 0);   // global_prefetch_b8
      __builtin_prefetch(W + k0 + 128, 0, 0);
      stage_load(X + k0 + 64, ldx, tx);
      stage_load(W + k0 + 64, ldw, tw);
    }
#pragma unroll
    for (int kk = 0; kk < 64; kk += 32) {
      v16bf a = load_frag_a(At + wave * 16 * LDA + kk, LDA);
      v16bf bcur = load_frag_b_nk(Wt + kk, LDA);
#pragma unroll
      for (int nt = 0; nt < 8; ++nt) {
        v16bf bnxt;
        if (nt < 7) bnxt = load_frag_b_nk(Wt + (nt + 1) * 16 * LDA + kk, LDA);
        acc[nt] = wmma_bf16(a, bcur, acc[nt]);
        if (nt < 7) bcur = bnxt;
      }
    }
    __syncthreads();
  }
}

// ===========================================================================
// Kernel 1: Q projection + per-head RMS norm + RoPE -> qn bf16 [B][NH][Q][HD]
// ===========================================================================
__global__ __launch_bounds__(256) void k_qproj(const float* __restrict__ hidden,
                                               const float* __restrict__ wq,
                                               const float* __restrict__ cosb,
                                               const float* __restrict__ sinb,
                                               const float* __restrict__ qnw,
                                               __bf16* __restrict__ qn) {
  __shared__ __bf16 At[128 * LDA];
  __shared__ __bf16 Wt[128 * LDA];
  const int h = blockIdx.x, b = blockIdx.y;
  const float* X = hidden + (size_t)b * QQ * HH;
  const float* W = wq + (size_t)h * HDD * HH;

  v8f acc[8];
#pragma unroll
  for (int t = 0; t < 8; ++t) acc[t] = zero8();
  gemm_mainloop(X, HH, W, HH, HH, At, Wt, acc);

  const int lane = threadIdx.x & 31, wave = threadIdx.x >> 5;
  const int ln = lane & 15, half = lane >> 4;
  float wcol[8];
#pragma unroll
  for (int t = 0; t < 8; ++t) wcol[t] = qnw[t * 16 + ln];

#pragma unroll
  for (int r = 0; r < 8; ++r) {
    float s = 0.0f;
#pragma unroll
    for (int t = 0; t < 8; ++t) s += acc[t][r] * acc[t][r];
#pragma unroll
    for (int off = 1; off < 16; off <<= 1) s += __shfl_xor(s, off, 32);
    const float inv = rsqrtf(s * (1.0f / HDD) + EPSF);
#pragma unroll
    for (int t = 0; t < 8; ++t) acc[t][r] = acc[t][r] * inv * wcol[t];
  }
  // RoPE: partner column (c+64)%128 lives in tile t^4, same reg/lane.
#pragma unroll
  for (int t = 0; t < 8; ++t) {
#pragma unroll
    for (int r = 0; r < 8; ++r) {
      const int qrow = wave * 16 + r + 8 * half;
      const int col = t * 16 + ln;
      const size_t cs = ((size_t)b * QQ + qrow) * HDD + col;
      const float c = cosb[cs], sn = sinb[cs];
      const float rot = (t < 4) ? -acc[t ^ 4][r] : acc[t ^ 4][r];
      const float o = acc[t][r] * c + rot * sn;
      qn[(((size_t)b * NHH + h) * QQ + qrow) * HDD + col] = (__bf16)o;
    }
  }
}

// ===========================================================================
// Kernel 2: K/V projections over concat(target, hidden) (S=4224 rows).
// ===========================================================================
__global__ __launch_bounds__(256) void k_kvproj(const float* __restrict__ hidden,
                                                const float* __restrict__ target,
                                                const float* __restrict__ wk,
                                                const float* __restrict__ wv,
                                                const float* __restrict__ cosb,
                                                const float* __restrict__ sinb,
                                                const float* __restrict__ knw,
                                                __bf16* __restrict__ kbuf,
                                                __bf16* __restrict__ vbuf) {
  __shared__ __bf16 At[128 * LDA];
  __shared__ __bf16 Wt[128 * LDA];
  const int mt = blockIdx.x;            // 0..32
  const int is_k = (blockIdx.y < 8);
  const int h = blockIdx.y & 7;
  const int b = blockIdx.z;
  const float* X = (mt < 32) ? (target + ((size_t)b * CTXN + (size_t)mt * 128) * HH)
                             : (hidden + (size_t)b * QQ * HH);
  const float* W = (is_k ? wk : wv) + (size_t)h * HDD * HH;

  v8f acc[8];
#pragma unroll
  for (int t = 0; t < 8; ++t) acc[t] = zero8();
  gemm_mainloop(X, HH, W, HH, HH, At, Wt, acc);

  const int lane = threadIdx.x & 31, wave = threadIdx.x >> 5;
  const int ln = lane & 15, half = lane >> 4;
  __bf16* dst = is_k ? kbuf : vbuf;

  if (is_k) {
    float wcol[8];
#pragma unroll
    for (int t = 0; t < 8; ++t) wcol[t] = knw[t * 16 + ln];
#pragma unroll
    for (int r = 0; r < 8; ++r) {
      float s = 0.0f;
#pragma unroll
      for (int t = 0; t < 8; ++t) s += acc[t][r] * acc[t][r];
#pragma unroll
      for (int off = 1; off < 16; off <<= 1) s += __shfl_xor(s, off, 32);
      const float inv = rsqrtf(s * (1.0f / HDD) + EPSF);
#pragma unroll
      for (int t = 0; t < 8; ++t) acc[t][r] = acc[t][r] * inv * wcol[t];
    }
  }
#pragma unroll
  for (int t = 0; t < 8; ++t) {
#pragma unroll
    for (int r = 0; r < 8; ++r) {
      const int lrow = wave * 16 + r + 8 * half;
      const int col = t * 16 + ln;
      float o = acc[t][r];
      if (is_k && mt == 32) {  // RoPE only on the appended hidden rows
        const size_t cs = ((size_t)b * QQ + lrow) * HDD + col;
        const float rot = (t < 4) ? -acc[t ^ 4][r] : acc[t ^ 4][r];
        o = o * cosb[cs] + rot * sinb[cs];
      }
      const int srow = mt * 128 + lrow;
      dst[(((size_t)b * NKVH + h) * SS + srow) * HDD + col] = (__bf16)o;
    }
  }
}

// ===========================================================================
// Kernel 3: GQA flash attention.
//   K: double-buffered via global_load_async_to_lds_b128 (ASYNCcnt).
//   V: DMA'd by the Tensor Data Mover (TENSORcnt), issued by wave 0.
//   P tile overlays the current K buffer (128*68 == 64*136 elements).
// ===========================================================================
__global__ __launch_bounds__(256) void k_attn(const __bf16* __restrict__ qn,
                                              const __bf16* __restrict__ kbuf,
                                              const __bf16* __restrict__ vbuf,
                                              float* __restrict__ attn_out) {
  __shared__ __bf16 ktile[2][KTT * KLD];  // 2 x 17408 B
  __shared__ __bf16 vtile[KTT * VLD];     // 16384 B
  const int h = blockIdx.x, b = blockIdx.y;
  const int h2 = h >> 1;  // GQA groups = 2
  const __bf16* qp = qn + (((size_t)b * NHH + h) * QQ) * HDD;
  const __bf16* kp = kbuf + (((size_t)b * NKVH + h2) * SS) * HDD;
  const __bf16* vp = vbuf + (((size_t)b * NKVH + h2) * SS) * HDD;
  const int tid = threadIdx.x, wave = tid >> 5, lane = tid & 31;
  const int ln = lane & 15, half = lane >> 4;

  // Preload this wave's 16 query rows as 4 A-fragments (HD = 4 x K32).
  v16bf qa[4];
#pragma unroll
  for (int kk = 0; kk < 4; ++kk) {
    const __bf16* row = qp + (size_t)(wave * 16 + ln) * HDD + kk * 32 + 8 * half;
    v16bf a;
#pragma unroll
    for (int j = 0; j < 8; ++j) {
      const int kb = (j < 4) ? (2 * j) : (2 * j + 8);
      bf16x2 p = *(const bf16x2*)(row + kb);
      a[2 * j] = p.x;
      a[2 * j + 1] = p.y;
    }
    qa[kk] = a;
  }

  float rmax[8], rsum[8];
  v8f out[8];
#pragma unroll
  for (int r = 0; r < 8; ++r) { rmax[r] = -3.0e38f; rsum[r] = 0.0f; }
#pragma unroll
  for (int t = 0; t < 8; ++t) out[t] = zero8();

  const float scaling = 0.08838834764831845f;  // HD^-0.5

  // Prologue: bring K tile 0 in.
  async_tile(kp, ktile[0], KLD);
  asm volatile("s_wait_asynccnt 0" ::: "memory");
  __syncthreads();

  const int NT = SS / KTT;  // 66
  for (int it = 0; it < NT; ++it) {
    const int cur = it & 1;
    // Kick off this tile's V DMA (TDM, wave 0) and next tile's K (async).
    if (wave == 0) tdm_load_tile(vp + (size_t)it * KTT * HDD, vtile);
    if (it + 1 < NT) async_tile(kp + (size_t)(it + 1) * KTT * HDD, ktile[cur ^ 1], KLD);

    // Scores: S(16q x 64keys) = Q * K^T from ktile[cur] ([key][hd] -> nk),
    // with next-fragment prefetch to overlap ds reads and WMMAs.
    v8f sc[4];
#pragma unroll
    for (int nt = 0; nt < 4; ++nt) sc[nt] = zero8();
#pragma unroll
    for (int kk = 0; kk < 4; ++kk) {
      v16bf bcur = load_frag_b_nk(ktile[cur] + kk * 32, KLD);
#pragma unroll
      for (int nt = 0; nt < 4; ++nt) {
        v16bf bnxt;
        if (nt < 3) bnxt = load_frag_b_nk(ktile[cur] + (nt + 1) * 16 * KLD + kk * 32, KLD);
        sc[nt] = wmma_bf16(qa[kk], bcur, sc[nt]);
        if (nt < 3) bcur = bnxt;
      }
    }

    // Online softmax (rows wave-local; reduce across the 16-lane half).
#pragma unroll
    for (int r = 0; r < 8; ++r) {
      float tmax = -3.0e38f;
#pragma unroll
      for (int nt = 0; nt < 4; ++nt) {
        const float v = sc[nt][r] * scaling;
        sc[nt][r] = v;
        tmax = fmaxf(tmax, v);
      }
#pragma unroll
      for (int off = 1; off < 16; off <<= 1)
        tmax = fmaxf(tmax, __shfl_xor(tmax, off, 32));
      const float mnew = fmaxf(rmax[r], tmax);
      const float alpha = __expf(rmax[r] - mnew);
      rmax[r] = mnew;
      float tsum = 0.0f;
#pragma unroll
      for (int nt = 0; nt < 4; ++nt) {
        const float p = __expf(sc[nt][r] - mnew);
        sc[nt][r] = p;
        tsum += p;
      }
#pragma unroll
      for (int off = 1; off < 16; off <<= 1) tsum += __shfl_xor(tsum, off, 32);
      rsum[r] = rsum[r] * alpha + tsum;
#pragma unroll
      for (int ht = 0; ht < 8; ++ht) out[ht][r] *= alpha;
    }

    // V(it) resident (TDM done) + all waves done reading ktile[cur].
    if (wave == 0) __builtin_amdgcn_s_wait_tensorcnt(0);
    __syncthreads();

    // Write P (wave-private rows) into the current K buffer.
    __bf16* pt = ktile[cur];
#pragma unroll
    for (int nt = 0; nt < 4; ++nt) {
#pragma unroll
      for (int r = 0; r < 8; ++r) {
        pt[(wave * 16 + r + 8 * half) * PLD + nt * 16 + ln] = (__bf16)sc[nt][r];
      }
    }

    // out += P(16x64) * V(64x128); vtile stored [key][hd] -> kn layout.
#pragma unroll
    for (int kk = 0; kk < 2; ++kk) {
      v16bf pa = load_frag_a(pt + wave * 16 * PLD + kk * 32, PLD);
      v16bf bcur = load_frag_b_kn(vtile + kk * 32 * VLD, VLD);
#pragma unroll
      for (int ht = 0; ht < 8; ++ht) {
        v16bf bnxt;
        if (ht < 7) bnxt = load_frag_b_kn(vtile + kk * 32 * VLD + (ht + 1) * 16, VLD);
        out[ht] = wmma_bf16(pa, bcur, out[ht]);
        if (ht < 7) bcur = bnxt;
      }
    }

    // K(it+1) resident; barrier before next iteration reuses the buffers.
    asm volatile("s_wait_asynccnt 0" ::: "memory");
    __syncthreads();
  }

  // Normalize and store: attn_out [B][Q][NH*HD] f32.
#pragma unroll
  for (int ht = 0; ht < 8; ++ht) {
#pragma unroll
    for (int r = 0; r < 8; ++r) {
      const int qrow = wave * 16 + r + 8 * half;
      attn_out[((size_t)b * QQ + qrow) * (NHH * HDD) + h * HDD + ht * 16 + ln] =
          out[ht][r] / rsum[r];
    }
  }
}

// ===========================================================================
// Kernel 4: output projection: d_out = attn_out @ wo.T
// ===========================================================================
__global__ __launch_bounds__(256) void k_oproj(const float* __restrict__ attn,
                                               const float* __restrict__ wo,
                                               float* __restrict__ outp) {
  __shared__ __bf16 At[128 * LDA];
  __shared__ __bf16 Wt[128 * LDA];
  const int ntile = blockIdx.x, b = blockIdx.y;
  const float* X = attn + (size_t)b * QQ * (NHH * HDD);
  const float* W = wo + (size_t)ntile * 128 * (NHH * HDD);

  v8f acc[8];
#pragma unroll
  for (int t = 0; t < 8; ++t) acc[t] = zero8();
  gemm_mainloop(X, NHH * HDD, W, NHH * HDD, NHH * HDD, At, Wt, acc);

  const int lane = threadIdx.x & 31, wave = threadIdx.x >> 5;
  const int ln = lane & 15, half = lane >> 4;
#pragma unroll
  for (int t = 0; t < 8; ++t) {
#pragma unroll
    for (int r = 0; r < 8; ++r) {
      const int qrow = wave * 16 + r + 8 * half;
      outp[((size_t)b * QQ + qrow) * HH + ntile * 128 + t * 16 + ln] = acc[t][r];
    }
  }
}

// ===========================================================================
extern "C" void kernel_launch(void* const* d_in, const int* in_sizes, int n_in,
                              void* d_out, int out_size, void* d_ws, size_t ws_size,
                              hipStream_t stream) {
  const float* hidden = (const float*)d_in[0];  // (B, Q, H)
  const float* target = (const float*)d_in[1];  // (B, CTX, H)
  const float* cosb   = (const float*)d_in[2];  // (B, Q, HD)
  const float* sinb   = (const float*)d_in[3];  // (B, Q, HD)
  const float* wq     = (const float*)d_in[4];  // (NH*HD, H)
  const float* wk     = (const float*)d_in[5];  // (NKV*HD, H)
  const float* wv     = (const float*)d_in[6];  // (NKV*HD, H)
  const float* wo     = (const float*)d_in[7];  // (H, NH*HD)
  const float* qnw    = (const float*)d_in[8];  // (HD,)
  const float* knw    = (const float*)d_in[9];  // (HD,)

  const size_t qn_elems = (size_t)Bb * NHH * QQ * HDD;    // 2,097,152
  const size_t kv_elems = (size_t)Bb * NKVH * SS * HDD;   // 34,603,008 each
  __bf16* qn   = (__bf16*)d_ws;
  __bf16* kbuf = qn + qn_elems;
  __bf16* vbuf = kbuf + kv_elems;
  float*  attn = (float*)(vbuf + kv_elems);               // 8 MB f32

  k_qproj<<<dim3(NHH, Bb), 256, 0, stream>>>(hidden, wq, cosb, sinb, qnw, qn);
  k_kvproj<<<dim3(SS / 128, 2 * NKVH, Bb), 256, 0, stream>>>(
      hidden, target, wk, wv, cosb, sinb, knw, kbuf, vbuf);
  k_attn<<<dim3(NHH, Bb), 256, 0, stream>>>(qn, kbuf, vbuf, attn);
  k_oproj<<<dim3(HH / 128, Bb), 256, 0, stream>>>(attn, wo, (float*)d_out);
}
